// MultiHeadAttention_13735305412662
// MI455X (gfx1250) — matrix operand order repaired and checked by reading, not compiled, not run on hardware
//
#include <hip/hip_runtime.h>
#include <hip/hip_bf16.h>

typedef __bf16 bf16;
typedef __attribute__((ext_vector_type(16))) __bf16 v16bf;
typedef __attribute__((ext_vector_type(8)))  __bf16 v8bf;
typedef __attribute__((ext_vector_type(8)))  float   v8f;
typedef __attribute__((ext_vector_type(4)))  float   v4f;
typedef __attribute__((ext_vector_type(4)))  unsigned uint4v;
typedef __attribute__((ext_vector_type(8)))  unsigned uint8v;

constexpr int BATCH = 4, SEQ = 1024, DMODEL = 1024, NH = 16, DHEAD = 64;

static __device__ __forceinline__ v16bf cat8(v8bf lo, v8bf hi) {
  return __builtin_shufflevector(lo, hi, 0,1,2,3,4,5,6,7,8,9,10,11,12,13,14,15);
}
static __device__ __forceinline__ v8bf cvt8(v4f a, v4f b) {
  v8bf r;
#pragma unroll
  for (int i = 0; i < 4; ++i) { r[i] = (bf16)a[i]; r[i + 4] = (bf16)b[i]; }
  return r;
}
static __device__ __forceinline__ v8f wmma_bf16(v16bf a, v16bf b, v8f c) {
  return __builtin_amdgcn_wmma_f32_16x16x32_bf16(false, a, false, b, (short)0, c,
                                                 false, false);
}

// ---------------------------------------------------------------------------
// Tensor Data Mover: issue a 2-D tile load (tile_d1 rows x tile_d0 elements,
// global row stride = stride0 elements) into LDS at lds_addr. D# per ISA
// CDNA5 ISA: group0 = {count/flags, lds_addr, global_addr, type=2},
// group1 = {data_size/pad, tensor dims, tile dims, dim0 stride}. Optional LDS
// row padding (pad_dwords DWORDs every 2^(pad_code+1) DWORDs) gives padded
// LDS row strides without a repack pass. Uniform operands -> SGPR groups.
// ---------------------------------------------------------------------------
static __device__ __forceinline__ void tdm_load_2d(
    const void* gsrc, unsigned lds_addr, unsigned data_size_code,
    unsigned tile_d0, unsigned tile_d1, unsigned stride0_elems,
    unsigned pad_code, unsigned pad_dwords)
{
  unsigned long long ga = (unsigned long long)(size_t)gsrc;
  uint4v g0;
  g0.x = 1u;                                   // count=1, user descriptor
  g0.y = lds_addr;                             // LDS byte address
  g0.z = (unsigned)ga;                         // global_addr[31:0]
  g0.w = (unsigned)(ga >> 32) | (2u << 30);    // global_addr[56:32] | type=2
  unsigned d0 = data_size_code << 16;          // data_size: 0=1B,1=2B,2=4B
  if (pad_dwords)
    d0 |= (1u << 20) | (pad_code << 22) | ((pad_dwords - 1u) << 25);
  const unsigned td0 = 1u << 30, td1 = 1u << 20;   // huge tensor dims: no OOB
  uint8v g1;
  g1[0] = d0;
  g1[1] = (td0 & 0xffffu) << 16;                       // tensor_dim0[15:0]
  g1[2] = (td0 >> 16) | ((td1 & 0xffffu) << 16);       // td0 hi | td1 lo
  g1[3] = (td1 >> 16) | (tile_d0 << 16);               // td1 hi | tile_dim0
  g1[4] = tile_d1 & 0xffffu;                           // tile_dim1 (tile_dim2=0)
  g1[5] = stride0_elems;                               // tensor_dim0_stride lo32
  g1[6] = 0u;
  g1[7] = 0u;
  asm volatile("tensor_load_to_lds %0, %1" :: "s"(g0), "s"(g1) : "memory");
}

static __device__ __forceinline__ void tdm_wait0() {
#if __has_builtin(__builtin_amdgcn_s_wait_tensorcnt)
  __builtin_amdgcn_s_wait_tensorcnt(0);
#else
  asm volatile("s_wait_tensorcnt 0x0" ::: "memory");
#endif
}

// ---------------------------------------------------------------------------
// GEMM: C[M][N] = A[M][K] @ W[N][K]^T + bias[N]
// Block = 256 threads (8 waves), block tile 128(M) x 64(N), K step 32.
// A and W tiles are DMA'd into LDS by the TDM (wave 0 issues descriptors,
// waits TENSORcnt, block barriers). Fragments are read straight from the
// padded raw tiles; f32 operands convert to bf16 in registers at feed time.
// ---------------------------------------------------------------------------
template <bool A_BF16, bool OUT_BF16>
__global__ __launch_bounds__(256)
void gemm_wmma_kernel(const void* __restrict__ Aptr, const float* __restrict__ W,
                      const float* __restrict__ bias, void* __restrict__ Cptr,
                      int M, int N, int K)
{
  constexpr int BM = 128, BN = 64, BK = 32;
  constexpr int AROWB = A_BF16 ? 80 : 144;    // TDM-padded LDS row stride (bytes)
  constexpr int WROWB = 144;                  // 32 f32 + 4 pad DWORDs
  __shared__ __align__(16) unsigned char Araw[BM * AROWB];
  __shared__ __align__(16) unsigned char Wraw[BN * WROWB];

  const int tid  = threadIdx.x;
  const int lane = tid & 31;
  const int wave = tid >> 5;
  const int m0 = blockIdx.y * BM;
  const int n0 = blockIdx.x * BN;

  const unsigned ldsA = (unsigned)(size_t)&Araw[0];
  const unsigned ldsW = (unsigned)(size_t)&Wraw[0];

  v8f acc[4];
#pragma unroll
  for (int t = 0; t < 4; ++t)
#pragma unroll
    for (int i = 0; i < 8; ++i) acc[t][i] = 0.0f;

  // A fragment: lane l<16 -> row l, K {0..7,16..23}; l>=16 -> K {8..15,24..31}
  const int mrow = wave * 16 + (lane & 15);
  const int ksel = (lane >> 4) * 8;
  const int kb   = (lane >> 4) * 16;

  for (int kc = 0; kc < K; kc += BK) {
    if (wave == 0) {
      tdm_load_2d(W + (size_t)n0 * K + kc, ldsW, /*f32*/2u, BK, BN,
                  (unsigned)K, /*32 dw rows*/4u, /*pad*/4u);
      if (A_BF16)
        tdm_load_2d((const bf16*)Aptr + (size_t)m0 * K + kc, ldsA, /*bf16*/1u,
                    BK, BM, (unsigned)K, /*16 dw rows*/3u, /*pad*/4u);
      else
        tdm_load_2d((const float*)Aptr + (size_t)m0 * K + kc, ldsA, /*f32*/2u,
                    BK, BM, (unsigned)K, 4u, 4u);
      tdm_wait0();
    }
    __syncthreads();

    v16bf afrag;
    const unsigned char* arow = &Araw[(size_t)mrow * AROWB];
    if (A_BF16) {
      afrag = cat8(*(const v8bf*)(arow + ksel * 2),
                   *(const v8bf*)(arow + ksel * 2 + 32));
    } else {
      v4f a0 = *(const v4f*)(arow + ksel * 4);
      v4f a1 = *(const v4f*)(arow + ksel * 4 + 16);
      v4f a2 = *(const v4f*)(arow + ksel * 4 + 64);
      v4f a3 = *(const v4f*)(arow + ksel * 4 + 80);
      afrag = cat8(cvt8(a0, a1), cvt8(a2, a3));
    }
#pragma unroll
    for (int nt = 0; nt < 4; ++nt) {
      const int ncol = nt * 16 + (lane & 15);
      const unsigned char* wrow = &Wraw[(size_t)ncol * WROWB + (size_t)ksel * 4];
      v4f b0 = *(const v4f*)(wrow);
      v4f b1 = *(const v4f*)(wrow + 16);
      v4f b2 = *(const v4f*)(wrow + 64);
      v4f b3 = *(const v4f*)(wrow + 80);
      acc[nt] = wmma_bf16(afrag, cat8(cvt8(b0, b1), cvt8(b2, b3)), acc[nt]);
    }
    __syncthreads();   // all reads done before next iteration's TDM overwrites
  }

  // C layout: VGPR r, lane l -> (M = r + 8*(l>>4), N = l&15)
  const int rowBase = m0 + wave * 16 + (lane >> 4) * 8;
  const int colLow  = lane & 15;
#pragma unroll
  for (int nt = 0; nt < 4; ++nt) {
    const int gcol = n0 + nt * 16 + colLow;
    const float bv = bias[gcol];
#pragma unroll
    for (int r = 0; r < 8; ++r) {
      const float  val = acc[nt][r] + bv;
      const size_t idx = (size_t)(rowBase + r) * N + gcol;
      if (OUT_BF16) ((bf16*)Cptr)[idx] = (bf16)val;
      else          ((float*)Cptr)[idx] = val;
    }
  }
}

// ---------------------------------------------------------------------------
// Scores: for each (b,h): S[sq][sk] = (q . k)/8 with causal mask -> -1e9.
// One wave per 16x16 tile; K = DHEAD = 64 => 2 WMMA. Tiles strictly above the
// diagonal are 100% masked: skip loads + WMMA, store the constant.
// ---------------------------------------------------------------------------
__global__ __launch_bounds__(256)
void scores_kernel(const bf16* __restrict__ q, const bf16* __restrict__ k,
                   float* __restrict__ attn)
{
  const int lane = threadIdx.x & 31;
  const int wave = threadIdx.x >> 5;
  const int tk = blockIdx.x * 8 + wave;   // 0..63
  const int tq = blockIdx.y;              // 0..63
  const int bh = blockIdx.z;              // 0..63
  const int b  = bh >> 4, h = bh & 15;

  float* base = attn + (size_t)bh * SEQ * SEQ;
  const int sk = tk * 16 + (lane & 15);

  if (tk > tq) {                          // fully masked tile
#pragma unroll
    for (int r = 0; r < 8; ++r) {
      const int sq = tq * 16 + r + (lane >> 4) * 8;
      base[(size_t)sq * SEQ + sk] = -1e9f;
    }
    return;
  }

  const bf16* qh = q + (size_t)b * SEQ * DMODEL + (size_t)h * DHEAD;
  const bf16* kh = k + (size_t)b * SEQ * DMODEL + (size_t)h * DHEAD;
  const int mr = tq * 16 + (lane & 15);

  v8f acc;
#pragma unroll
  for (int i = 0; i < 8; ++i) acc[i] = 0.0f;

#pragma unroll
  for (int kc = 0; kc < DHEAD; kc += 32) {
    const int ka = kc + ((lane >> 4) ? 8 : 0);
    const bf16* arow = qh + (size_t)mr * DMODEL;
    v16bf afrag = cat8(*(const v8bf*)(arow + ka),
                       *(const v8bf*)(arow + ka + 16));
    const int kb = kc + (lane >> 4) * 8;
    const bf16* brow = kh + (size_t)sk * DMODEL;
    v16bf bfrag = cat8(*(const v8bf*)(brow + kb),
                       *(const v8bf*)(brow + kb + 16));
    acc = wmma_bf16(afrag, bfrag, acc);
  }

  const float scale = 0.125f;             // 1/sqrt(64)
#pragma unroll
  for (int r = 0; r < 8; ++r) {
    const int sq = tq * 16 + r + (lane >> 4) * 8;
    float v = acc[r] * scale;
    if (sk > sq) v = -1e9f;               // diagonal tile partial mask
    base[(size_t)sq * SEQ + sk] = v;
  }
}

// ---------------------------------------------------------------------------
// Row softmax in place over attn rows of length SEQ. One 256-thread block/row.
// ---------------------------------------------------------------------------
__global__ __launch_bounds__(256)
void softmax_kernel(float* __restrict__ attn)
{
  const size_t row = blockIdx.x;
  float* p = attn + row * SEQ;
  const int tid = threadIdx.x;
  __shared__ float red[256];

  float vals[4];
  float m = -3.4e38f;
#pragma unroll
  for (int j = 0; j < 4; ++j) {
    vals[j] = p[tid + j * 256];
    m = fmaxf(m, vals[j]);
  }
  red[tid] = m;
  __syncthreads();
  for (int s = 128; s > 0; s >>= 1) {
    if (tid < s) red[tid] = fmaxf(red[tid], red[tid + s]);
    __syncthreads();
  }
  m = red[0];
  __syncthreads();

  float sum = 0.0f;
#pragma unroll
  for (int j = 0; j < 4; ++j) {
    vals[j] = __expf(vals[j] - m);
    sum += vals[j];
  }
  red[tid] = sum;
  __syncthreads();
  for (int s = 128; s > 0; s >>= 1) {
    if (tid < s) red[tid] += red[tid + s];
    __syncthreads();
  }
  const float inv = 1.0f / red[0];
#pragma unroll
  for (int j = 0; j < 4; ++j) p[tid + j * 256] = vals[j] * inv;
}

// ---------------------------------------------------------------------------
// x = P @ V per (b,h): M=SEQ, K=SEQ, N=DHEAD. Block = 8 waves -> 32x64 output.
// V k-tile staged *transposed* into LDS (TDM can't transpose); P converted
// f32->bf16 at fragment-build time.
// ---------------------------------------------------------------------------
__global__ __launch_bounds__(256)
void pv_kernel(const float* __restrict__ attn, const bf16* __restrict__ v,
               bf16* __restrict__ x)
{
  constexpr int BK = 32, LDS_STRIDE = BK + 8;
  __shared__ __align__(16) bf16 Vt[DHEAD][LDS_STRIDE];

  const int tid  = threadIdx.x;
  const int lane = tid & 31;
  const int wave = tid >> 5;
  const int m0 = blockIdx.x * 32;
  const int bh = blockIdx.y;
  const int b  = bh >> 4, h = bh & 15;

  const float* P  = attn + (size_t)bh * SEQ * SEQ;
  const bf16*  vh = v + (size_t)b * SEQ * DMODEL + (size_t)h * DHEAD;

  const int mt = wave >> 2;               // 0..1 (M tile)
  const int nt = wave & 3;                // 0..3 (N tile)
  const int mr = m0 + mt * 16 + (lane & 15);

  v8f acc;
#pragma unroll
  for (int i = 0; i < 8; ++i) acc[i] = 0.0f;

  for (int kc = 0; kc < SEQ; kc += BK) {
    { // stage V^T tile: Vt[dk][kk]
      const int kk  = tid >> 3;           // 0..31
      const int dko = (tid & 7) * 8;      // 0..56
      v8bf vv = *(const v8bf*)(vh + (size_t)(kc + kk) * DMODEL + dko);
#pragma unroll
      for (int j = 0; j < 8; ++j) Vt[dko + j][kk] = vv[j];
    }
    __syncthreads();

    const float* prow = P + (size_t)mr * SEQ;
    const int ka = kc + ((lane >> 4) ? 8 : 0);
    v4f p0 = *(const v4f*)(prow + ka);
    v4f p1 = *(const v4f*)(prow + ka + 4);
    v4f p2 = *(const v4f*)(prow + ka + 16);
    v4f p3 = *(const v4f*)(prow + ka + 20);
    v16bf afrag = cat8(cvt8(p0, p1), cvt8(p2, p3));

    const int ncol = nt * 16 + (lane & 15);
    const int kb   = (lane >> 4) * 8;
    v16bf bfrag = cat8(*(const v8bf*)&Vt[ncol][kb],
                       *(const v8bf*)&Vt[ncol][kb + 16]);
    acc = wmma_bf16(afrag, bfrag, acc);
    __syncthreads();
  }

  bf16* xh = x + (size_t)b * SEQ * DMODEL + (size_t)h * DHEAD;
  const int col = nt * 16 + (lane & 15);
  const int rb  = m0 + mt * 16 + (lane >> 4) * 8;
#pragma unroll
  for (int r = 0; r < 8; ++r)
    xh[(size_t)(rb + r) * DMODEL + col] = (bf16)acc[r];
}

// ---------------------------------------------------------------------------
extern "C" void kernel_launch(void* const* d_in, const int* in_sizes, int n_in,
                              void* d_out, int out_size, void* d_ws, size_t ws_size,
                              hipStream_t stream)
{
  (void)in_sizes; (void)n_in; (void)out_size; (void)ws_size;

  const float* Q  = (const float*)d_in[0];
  const float* K  = (const float*)d_in[1];
  const float* V  = (const float*)d_in[2];
  // d_in[3] = mask (causal; computed analytically, unused)
  const float* Wq = (const float*)d_in[4];
  const float* bq = (const float*)d_in[5];
  const float* Wk = (const float*)d_in[6];
  const float* bk = (const float*)d_in[7];
  const float* Wv = (const float*)d_in[8];
  const float* bv = (const float*)d_in[9];
  const float* Wo = (const float*)d_in[10];
  const float* bo = (const float*)d_in[11];

  float* out  = (float*)d_out;                       // [B,S,D]
  float* attn = out + (size_t)BATCH * SEQ * DMODEL;  // [B,H,S,S]

  const size_t nTok = (size_t)BATCH * SEQ * DMODEL;  // 4M elems
  bf16* q_ws = (bf16*)d_ws;
  bf16* k_ws = q_ws + nTok;
  bf16* v_ws = k_ws + nTok;
  bf16* x_ws = q_ws;                                 // reuse q slot after scores

  const int M = BATCH * SEQ;                         // 4096
  dim3 gemmGrid(DMODEL / 64, M / 128);

  gemm_wmma_kernel<false, true><<<gemmGrid, 256, 0, stream>>>(
      Q, Wq, bq, q_ws, M, DMODEL, DMODEL);
  gemm_wmma_kernel<false, true><<<gemmGrid, 256, 0, stream>>>(
      K, Wk, bk, k_ws, M, DMODEL, DMODEL);
  gemm_wmma_kernel<false, true><<<gemmGrid, 256, 0, stream>>>(
      V, Wv, bv, v_ws, M, DMODEL, DMODEL);

  scores_kernel<<<dim3(SEQ / 16 / 8, SEQ / 16, BATCH * NH), 256, 0, stream>>>(
      q_ws, k_ws, attn);

  softmax_kernel<<<dim3(BATCH * NH * SEQ), 256, 0, stream>>>(attn);

  pv_kernel<<<dim3(SEQ / 32, BATCH * NH), 256, 0, stream>>>(attn, v_ws, x_ws);

  gemm_wmma_kernel<true, false><<<gemmGrid, 256, 0, stream>>>(
      x_ws, Wo, bo, out, M, DMODEL, DMODEL);
}
